// SpatialDCNFusion_14998025797719
// MI455X (gfx1250) — compile-verified, matrix operand order checked by reading, not run on hardware
//
#include <hip/hip_runtime.h>
#include <hip/hip_bf16.h>

// ---------------------------------------------------------------------------
// Problem constants
// ---------------------------------------------------------------------------
#define B_    2
#define H_    128
#define W_    128
#define CL_   128    // CIN_LOW
#define CH_   256    // CIN_HIGH (= channels of x_merge)
#define CO_   128    // COUT
#define HW_   (H_*W_)
#define NPIX_ (B_*HW_)          // 32768 pixels

typedef __attribute__((ext_vector_type(16))) __bf16 v16bf;
typedef __attribute__((ext_vector_type(8)))  __bf16 v8bf;
typedef __attribute__((ext_vector_type(8)))  float  v8f;

// ---------------------------------------------------------------------------
// bf16 <-> f32: native casts so the backend can use v_cvt_pk_bf16_f32 and
// fold up-converts into v_fma_mix_f32_bf16.
// ---------------------------------------------------------------------------
__device__ __forceinline__ __bf16 f2bf(float f) { return (__bf16)f; }
__device__ __forceinline__ float  bf2f(__bf16 b) { return (float)b; }
__device__ __forceinline__ __bf16 bf_zero() {
    unsigned short z = 0;
    return __builtin_bit_cast(__bf16, z);
}

__device__ __forceinline__ v8f wmma_bf16(v16bf a, v16bf b, v8f c) {
    return __builtin_amdgcn_wmma_f32_16x16x32_bf16(
        /*neg_a=*/false, a, /*neg_b=*/false, b,
        /*c_mod=*/(short)0, c, /*reuse_a=*/false, /*reuse_b=*/false);
}

// load 16 contiguous bf16 into fragment element order 0..15
__device__ __forceinline__ v16bf load16(const __bf16* __restrict__ p) {
    v8bf lo = *(const v8bf*)p;
    v8bf hi = *(const v8bf*)(p + 8);
    v16bf r;
#pragma unroll
    for (int i = 0; i < 8; ++i) { r[i] = lo[i]; r[i + 8] = hi[i]; }
    return r;
}

// ---------------------------------------------------------------------------
// Weight packing: w[CO][CIN][3][3] f32 -> fragment-ordered bf16
// layout: [kslice][ntile(8)][lane(32)][elem(16)], k = tap*CIN + c,
// per-fragment element: k_in_slice = (e&7) + 16*(e>>3) + 8*(lane>>4), n = nt*16 + (lane&15)
// ---------------------------------------------------------------------------
__global__ void pack_weights(const float* __restrict__ w, __bf16* __restrict__ wp, int CIN) {
    const int id = blockIdx.x * blockDim.x + threadIdx.x;
    const int total = 9 * CIN * 128;
    if (id >= total) return;
    const int e    = id & 15;
    const int lane = (id >> 4) & 31;
    const int nt   = (id >> 9) & 7;
    const int ks   = id >> 12;
    const int k = ks * 32 + ((e & 7) + 16 * (e >> 3) + 8 * (lane >> 4));
    const int n = nt * 16 + (lane & 15);
    const int tap = k / CIN;
    const int c   = k % CIN;
    wp[id] = f2bf(w[(size_t)(n * CIN + c) * 9 + tap]);
}

// NCHW f32 -> NHWC bf16
__global__ void nchw_to_nhwc(const float* __restrict__ src, __bf16* __restrict__ dst, int C) {
    const int id = blockIdx.x * blockDim.x + threadIdx.x;
    if (id >= B_ * HW_ * C) return;
    const int hw = id % HW_;
    const int c  = (id / HW_) % C;
    const int b  = id / (HW_ * C);
    dst[((size_t)(b * HW_ + hw)) * C + c] = f2bf(src[id]);
}

// bilinear x2 upsample (half-pixel centers, edge clamp): NCHW f32 -> NHWC bf16
__global__ void upsample2x(const float* __restrict__ src, __bf16* __restrict__ dst) {
    const int id = blockIdx.x * blockDim.x + threadIdx.x;
    if (id >= B_ * CH_ * HW_) return;
    const int xo = id % W_;
    const int yo = (id / W_) % H_;
    const int c  = (id / HW_) % CH_;
    const int b  = id / (HW_ * CH_);
    const int HS = H_ / 2, WS = W_ / 2;
    const float fy = (yo + 0.5f) * 0.5f - 0.5f;
    const float fx = (xo + 0.5f) * 0.5f - 0.5f;
    const float y0f = floorf(fy), x0f = floorf(fx);
    const float wy = fy - y0f, wx = fx - x0f;
    int y0 = (int)y0f, x0 = (int)x0f;
    int y0c = y0 < 0 ? 0 : (y0 > HS - 1 ? HS - 1 : y0);
    int y1c = y0 + 1 < 0 ? 0 : (y0 + 1 > HS - 1 ? HS - 1 : y0 + 1);
    int x0c = x0 < 0 ? 0 : (x0 > WS - 1 ? WS - 1 : x0);
    int x1c = x0 + 1 < 0 ? 0 : (x0 + 1 > WS - 1 ? WS - 1 : x0 + 1);
    const float* sp = src + (size_t)(b * CH_ + c) * HS * WS;
    const float v00 = sp[y0c * WS + x0c];
    const float v01 = sp[y0c * WS + x1c];
    const float v10 = sp[y1c * WS + x0c];
    const float v11 = sp[y1c * WS + x1c];
    const float v = (1.f - wy) * ((1.f - wx) * v00 + wx * v01)
                  +        wy  * ((1.f - wx) * v10 + wx * v11);
    dst[((size_t)(b * HW_ + yo * W_ + xo)) * CH_ + c] = f2bf(v);
}

// ---------------------------------------------------------------------------
// 3x3 conv (implicit GEMM, WMMA bf16) + BN + ReLU, writes into x_merge slice.
// Wave: 32 consecutive pixels (2 M-subtiles) x 128 out channels (8 N-tiles).
// ---------------------------------------------------------------------------
template <int CIN>
__global__ void __launch_bounds__(256)
conv3x3_wmma(const __bf16* __restrict__ xin, const __bf16* __restrict__ wpack,
             const float* __restrict__ gg, const float* __restrict__ bb,
             const float* __restrict__ mm, const float* __restrict__ vv,
             __bf16* __restrict__ xm, int chanoff) {
    const int lane = threadIdx.x & 31;
    const int lm = lane & 15, lh = lane >> 4;
    const int wid = blockIdx.x * (blockDim.x >> 5) + (threadIdx.x >> 5);
    const int p0 = wid * 32;
    const int bimg = p0 / HW_;
    const int rem  = p0 % HW_;
    const int y    = rem / W_;
    const int x0   = rem % W_;

    v8f acc[2][8];
#pragma unroll
    for (int mi = 0; mi < 2; ++mi)
#pragma unroll
        for (int nt = 0; nt < 8; ++nt)
#pragma unroll
            for (int r = 0; r < 8; ++r) acc[mi][nt][r] = 0.f;

    for (int tap = 0; tap < 9; ++tap) {
        const int ky = tap / 3, kx = tap % 3;
        const int yy = y + ky - 1;
        const bool rowok = (yy >= 0) && (yy < H_);
        const __bf16* rowp = xin + (size_t)((bimg * H_ + yy) * W_) * CIN;
        for (int cs = 0; cs < CIN / 32; ++cs) {
            const int c0 = cs * 32;
            v16bf afrag[2];
#pragma unroll
            for (int mi = 0; mi < 2; ++mi) {
                const int xx = x0 + mi * 16 + lm + kx - 1;
                v16bf a;
                if (rowok && xx >= 0 && xx < W_) {
                    const __bf16* p = rowp + (size_t)xx * CIN + c0;
                    v8bf lo = *(const v8bf*)(p + 8 * lh);
                    v8bf hi = *(const v8bf*)(p + 16 + 8 * lh);
#pragma unroll
                    for (int i = 0; i < 8; ++i) { a[i] = lo[i]; a[i + 8] = hi[i]; }
                } else {
#pragma unroll
                    for (int i = 0; i < 16; ++i) a[i] = bf_zero();
                }
                afrag[mi] = a;
            }
            const __bf16* wp = wpack + (size_t)(tap * (CIN / 32) + cs) * 8 * 32 * 16;
#pragma unroll
            for (int nt = 0; nt < 8; ++nt) {
                v16bf bm = load16(wp + ((size_t)nt * 32 + lane) * 16);
                acc[0][nt] = wmma_bf16(afrag[0], bm, acc[0][nt]);
                acc[1][nt] = wmma_bf16(afrag[1], bm, acc[1][nt]);
            }
        }
    }

    // BN + ReLU epilogue -> x_merge NHWC bf16 (channel slice [chanoff, chanoff+128))
#pragma unroll
    for (int nt = 0; nt < 8; ++nt) {
        const int co = nt * 16 + lm;
        const float s = gg[co] * rsqrtf(vv[co] + 1e-5f);
        const float t = bb[co] - mm[co] * s;
#pragma unroll
        for (int mi = 0; mi < 2; ++mi) {
#pragma unroll
            for (int r = 0; r < 8; ++r) {
                const int pix = p0 + mi * 16 + r + 8 * lh;
                float vo = acc[mi][nt][r] * s + t;
                vo = vo > 0.f ? vo : 0.f;
                xm[(size_t)pix * CH_ + chanoff + co] = f2bf(vo);
            }
        }
    }
}

// ---------------------------------------------------------------------------
// 1x1 conv -> 18 offsets + 9 sigmoid masks per pixel ([NPIX][27] f32)
// ---------------------------------------------------------------------------
__global__ void __launch_bounds__(256)
offmask_kernel(const __bf16* __restrict__ xm, const float* __restrict__ woff,
               const float* __restrict__ boff, const float* __restrict__ wmask,
               const float* __restrict__ bmask, float* __restrict__ om) {
    __shared__ float wsh[27 * 256];
    for (int i = threadIdx.x; i < 27 * 256; i += blockDim.x)
        wsh[i] = (i < 18 * 256) ? woff[i] : wmask[i - 18 * 256];
    __syncthreads();

    const int p = blockIdx.x * blockDim.x + threadIdx.x;
    float acc[27];
#pragma unroll
    for (int o = 0; o < 27; ++o) acc[o] = 0.f;
    const __bf16* xp = xm + (size_t)p * CH_;
    for (int c8 = 0; c8 < CH_ / 8; ++c8) {
        v8bf v = *(const v8bf*)(xp + c8 * 8);
#pragma unroll
        for (int i = 0; i < 8; ++i) {
            const float xc = bf2f(v[i]);
            const int c = c8 * 8 + i;
#pragma unroll
            for (int o = 0; o < 27; ++o) acc[o] += xc * wsh[o * 256 + c];
        }
    }
    float* op = om + (size_t)p * 27;
#pragma unroll
    for (int o = 0; o < 18; ++o) op[o] = acc[o] + boff[o];
#pragma unroll
    for (int j = 0; j < 9; ++j) {
        const float z = acc[18 + j] + bmask[j];
        op[18 + j] = 1.f / (1.f + expf(-z));
    }
}

// ---------------------------------------------------------------------------
// Deformable 3x3 conv: bilinear-gather A fragments on the fly, WMMA GEMM
// (K = 9*256 = 2304), ReLU, NCHW f32 store. Wave: 16 pixels x 128 out ch.
// ---------------------------------------------------------------------------
__global__ void __launch_bounds__(256)
deform_wmma(const __bf16* __restrict__ xm, const float* __restrict__ om,
            const __bf16* __restrict__ wpack, float* __restrict__ out) {
    const int lane = threadIdx.x & 31;
    const int lm = lane & 15, lh = lane >> 4;
    const int wid = blockIdx.x * (blockDim.x >> 5) + (threadIdx.x >> 5);
    const int p0 = wid * 16;
    const int p  = p0 + lm;
    const int bimg = p / HW_;
    const int rem  = p % HW_;
    const int y = rem / W_, x = rem % W_;
    const float* omp = om + (size_t)p * 27;

    v8f acc[8];
#pragma unroll
    for (int nt = 0; nt < 8; ++nt)
#pragma unroll
        for (int r = 0; r < 8; ++r) acc[nt][r] = 0.f;

    for (int tap = 0; tap < 9; ++tap) {
        const float dy = omp[2 * tap];
        const float dx = omp[2 * tap + 1];
        const float mk = omp[18 + tap];
        const float ys = (float)(y + tap / 3 - 1) + dy;
        const float xs = (float)(x + tap % 3 - 1) + dx;
        const float y0f = floorf(ys), x0f = floorf(xs);
        const int iy0 = (int)y0f, ix0 = (int)x0f;
        const float wy = ys - y0f, wx = xs - x0f;
        float cw[4];
        cw[0] = (1.f - wy) * (1.f - wx) * mk;
        cw[1] = (1.f - wy) * wx * mk;
        cw[2] = wy * (1.f - wx) * mk;
        cw[3] = wy * wx * mk;
        const __bf16* cp[4];
#pragma unroll
        for (int cr = 0; cr < 4; ++cr) {
            const int yi = iy0 + (cr >> 1);
            const int xi = ix0 + (cr & 1);
            const bool vld = (yi >= 0) && (yi < H_) && (xi >= 0) && (xi < W_);
            const int yc = yi < 0 ? 0 : (yi > H_ - 1 ? H_ - 1 : yi);
            const int xc = xi < 0 ? 0 : (xi > W_ - 1 ? W_ - 1 : xi);
            cp[cr] = xm + (size_t)((bimg * H_ + yc) * W_ + xc) * CH_;
            if (!vld) cw[cr] = 0.f;
        }
        for (int cs = 0; cs < CH_ / 32; ++cs) {
            const int c0 = cs * 32;
            float f[16];
#pragma unroll
            for (int i = 0; i < 16; ++i) f[i] = 0.f;
#pragma unroll
            for (int cr = 0; cr < 4; ++cr) {
                const __bf16* q = cp[cr] + c0;
                v8bf lo = *(const v8bf*)(q + 8 * lh);
                v8bf hi = *(const v8bf*)(q + 16 + 8 * lh);
                const float w = cw[cr];
#pragma unroll
                for (int i = 0; i < 8; ++i) {
                    f[i]     += w * bf2f(lo[i]);      // -> v_fma_mix_f32_bf16
                    f[i + 8] += w * bf2f(hi[i]);
                }
            }
            v16bf a;
#pragma unroll
            for (int i = 0; i < 16; ++i) a[i] = f2bf(f[i]);  // -> v_cvt_pk_bf16_f32
            const __bf16* wp = wpack + (size_t)(tap * (CH_ / 32) + cs) * 8 * 32 * 16;
#pragma unroll
            for (int nt = 0; nt < 8; ++nt) {
                v16bf bm = load16(wp + ((size_t)nt * 32 + lane) * 16);
                acc[nt] = wmma_bf16(a, bm, acc[nt]);
            }
        }
    }

    // ReLU + NCHW f32 store
#pragma unroll
    for (int nt = 0; nt < 8; ++nt) {
        const int co = nt * 16 + lm;
#pragma unroll
        for (int r = 0; r < 8; ++r) {
            const int pix = p0 + r + 8 * lh;
            const int bo = pix / HW_;
            const int rr = pix % HW_;
            float vo = acc[nt][r];
            vo = vo > 0.f ? vo : 0.f;
            out[(size_t)(bo * CO_ + co) * HW_ + rr] = vo;
        }
    }
}

// ---------------------------------------------------------------------------
// Workspace layout (bytes)
// ---------------------------------------------------------------------------
#define XL_OFF   ((size_t)0)                               // NPIX*128 bf16 =  8 MB
#define XUP_OFF  (XL_OFF  + (size_t)NPIX_ * CL_ * 2)       // NPIX*256 bf16 = 16 MB
#define XM_OFF   (XUP_OFF + (size_t)NPIX_ * CH_ * 2)       // NPIX*256 bf16 = 16 MB
#define OM_OFF   (XM_OFF  + (size_t)NPIX_ * CH_ * 2)       // NPIX*27  f32  = 3.4 MB
#define WL_OFF   (OM_OFF  + (size_t)NPIX_ * 27 * 4)        // 1152*128 bf16
#define WH_OFF   (WL_OFF  + (size_t)9 * CL_ * CO_ * 2)     // 2304*128 bf16
#define WC_OFF   (WH_OFF  + (size_t)9 * CH_ * CO_ * 2)     // 2304*128 bf16

extern "C" void kernel_launch(void* const* d_in, const int* in_sizes, int n_in,
                              void* d_out, int out_size, void* d_ws, size_t ws_size,
                              hipStream_t stream) {
    const float* x_low  = (const float*)d_in[0];
    const float* x_high = (const float*)d_in[1];
    const float* w_low  = (const float*)d_in[2];
    const float* g_low  = (const float*)d_in[3];
    const float* b_low  = (const float*)d_in[4];
    const float* m_low  = (const float*)d_in[5];
    const float* v_low  = (const float*)d_in[6];
    const float* w_high = (const float*)d_in[7];
    const float* g_high = (const float*)d_in[8];
    const float* b_high = (const float*)d_in[9];
    const float* m_high = (const float*)d_in[10];
    const float* v_high = (const float*)d_in[11];
    const float* w_off  = (const float*)d_in[12];
    const float* b_off  = (const float*)d_in[13];
    const float* w_mask = (const float*)d_in[14];
    const float* b_mask = (const float*)d_in[15];
    const float* w_conv = (const float*)d_in[16];
    float* out = (float*)d_out;

    char* ws = (char*)d_ws;
    __bf16* xl  = (__bf16*)(ws + XL_OFF);
    __bf16* xup = (__bf16*)(ws + XUP_OFF);
    __bf16* xmg = (__bf16*)(ws + XM_OFF);
    float*  om  = (float*)(ws + OM_OFF);
    __bf16* wlp = (__bf16*)(ws + WL_OFF);
    __bf16* whp = (__bf16*)(ws + WH_OFF);
    __bf16* wcp = (__bf16*)(ws + WC_OFF);

    // weight packing (tiny)
    pack_weights<<<(9 * CL_ * CO_ + 255) / 256, 256, 0, stream>>>(w_low,  wlp, CL_);
    pack_weights<<<(9 * CH_ * CO_ + 255) / 256, 256, 0, stream>>>(w_high, whp, CH_);
    pack_weights<<<(9 * CH_ * CO_ + 255) / 256, 256, 0, stream>>>(w_conv, wcp, CH_);

    // activation layout conversion + upsample
    nchw_to_nhwc<<<(B_ * CL_ * HW_ + 255) / 256, 256, 0, stream>>>(x_low, xl, CL_);
    upsample2x<<<(B_ * CH_ * HW_ + 255) / 256, 256, 0, stream>>>(x_high, xup);

    // the two 3x3 convs (+BN+ReLU), fused concat into x_merge
    conv3x3_wmma<CL_><<<NPIX_ / 32 / 8, 256, 0, stream>>>(xl, wlp, g_low, b_low, m_low, v_low, xmg, 0);
    conv3x3_wmma<CH_><<<NPIX_ / 32 / 8, 256, 0, stream>>>(xup, whp, g_high, b_high, m_high, v_high, xmg, CL_);

    // offsets + masks (1x1 conv, sigmoid fused)
    offmask_kernel<<<NPIX_ / 256, 256, 0, stream>>>(xmg, w_off, b_off, w_mask, b_mask, om);

    // deformable conv (gather + WMMA GEMM + ReLU)
    deform_wmma<<<NPIX_ / 16 / 8, 256, 0, stream>>>(xmg, om, wcp, out);
}
